// MultiHeadAttention_2929167696619
// MI455X (gfx1250) — matrix operand order repaired and checked by reading, not compiled, not run on hardware
//
#include <hip/hip_runtime.h>
#include <hip/hip_bf16.h>
#include <math.h>

// ---------------------------------------------------------------------------
// Types for CDNA5 WMMA (wave32): A/B = 16 x bf16 per lane, C/D = 8 x f32.
// ---------------------------------------------------------------------------
typedef __attribute__((ext_vector_type(16))) __bf16        v16bf;
typedef __attribute__((ext_vector_type(8)))  float         v8f;
typedef __attribute__((ext_vector_type(8)))  unsigned short u16x8;
typedef __attribute__((ext_vector_type(4)))  unsigned short u16x4;

union FragAB { u16x8 h[2]; v16bf v; };   // 32 bytes

__device__ __forceinline__ unsigned short f2bf(float f) {
  unsigned u = __builtin_bit_cast(unsigned, f);
  return (unsigned short)((u + 0x7FFFu + ((u >> 16) & 1u)) >> 16);  // RNE
}

// LDS tile stride: 32 data + 8 pad bf16 => 80B rows; the 16-lane b128
// fragment reads land on 16 distinct 4-bank groups (no conflicts).
#define LDS_STRIDE 40

// ---------------------------------------------------------------------------
// Generic tiled WMMA GEMM:  C[m,n] = scale * sum_k A[m,k] * B[n,k]
//   A: bf16 row-major [M,lda]   (AMODE 0)  or fp32 row-major (AMODE 1)
//   B: bf16 row-major [N,ldb]   (i.e. K contiguous -> matches B fragment)
//   blockIdx.z selects a (batch,head) slice via aZ/bZ/dZ element strides.
// SMODE: 0 = head_out scatter bf16  [(z>>4)*2048+m, (z&15)*64+n] stride 1024
//        1 = Q/K scatter   bf16     [b,h,q,d] from (m=token, n=h*64+d)
//        2 = V^T scatter   bf16     [b,h,d,q]
//        3 = fp32 row-major          D[z*dZ + m*ldd + n]
// ---------------------------------------------------------------------------
struct GemmP {
  const void* A;
  const unsigned short* B;
  void* D;
  int M, N, K;
  int lda, ldb, ldd;
  long long aZ, bZ, dZ;
  float scale;
};

template <int AMODE, int SMODE>
__global__ __launch_bounds__(256) void gemm_wmma_bf16(GemmP p) {
  __shared__ unsigned short As[128 * LDS_STRIDE];
  __shared__ unsigned short Bs[128 * LDS_STRIDE];

  const int tid  = threadIdx.x;
  const int lane = tid & 31;
  const int wave = tid >> 5;
  const int wm   = wave & 3;    // 4 wave-rows  * 32 = 128
  const int wn   = wave >> 2;   // 2 wave-cols  * 64 = 128
  const int m0   = blockIdx.y * 128;
  const int n0   = blockIdx.x * 128;
  const int z    = blockIdx.z;

  const unsigned short* Bz  = p.B + (long long)z * p.bZ;
  const unsigned short* Abf = (const unsigned short*)p.A + (long long)z * p.aZ;
  const float*          Af  = (const float*)p.A + (long long)z * p.aZ;

  v8f acc[2][4];
#pragma unroll
  for (int i = 0; i < 2; ++i)
#pragma unroll
    for (int j = 0; j < 4; ++j)
#pragma unroll
      for (int r = 0; r < 8; ++r) acc[i][j][r] = 0.0f;

  for (int k0 = 0; k0 < p.K; k0 += 32) {
    // ---- stage 128x32 A tile and 128x32 B tile into LDS -------------------
#pragma unroll
    for (int half = 0; half < 2; ++half) {
      const int li  = (tid + half * 256) * 8;   // 0..4088, 8 bf16 per thread
      const int row = li >> 5;
      const int col = li & 31;

      if constexpr (AMODE == 0) {
        const u16x8* src =
            (const u16x8*)(Abf + (long long)(m0 + row) * p.lda + k0 + col);
        *(u16x8*)(As + row * LDS_STRIDE + col) = *src;
      } else {
        const float4* src =
            (const float4*)(Af + (long long)(m0 + row) * p.lda + k0 + col);
        float4 f0 = src[0], f1 = src[1];
        u16x8 v;
        v[0] = f2bf(f0.x); v[1] = f2bf(f0.y); v[2] = f2bf(f0.z); v[3] = f2bf(f0.w);
        v[4] = f2bf(f1.x); v[5] = f2bf(f1.y); v[6] = f2bf(f1.z); v[7] = f2bf(f1.w);
        *(u16x8*)(As + row * LDS_STRIDE + col) = v;
      }

      int grow = n0 + row;
      if (grow > p.N - 1) grow = p.N - 1;    // clamp when tile overhangs N
      const u16x8* bsrc =
          (const u16x8*)(Bz + (long long)grow * p.ldb + k0 + col);
      *(u16x8*)(Bs + row * LDS_STRIDE + col) = *bsrc;
    }
    __syncthreads();

    // ---- load fragments per ISA §7.12.2 layouts ---------------------------
    FragAB a[2], b[4];
#pragma unroll
    for (int mi = 0; mi < 2; ++mi) {
      const int r  = wm * 32 + mi * 16 + (lane & 15);
      const int kb = (lane >> 4) * 8;      // lanes 0-15: K 0-7/16-23; 16-31: 8-15/24-31
      a[mi].h[0] = *(const u16x8*)(As + r * LDS_STRIDE + kb);
      a[mi].h[1] = *(const u16x8*)(As + r * LDS_STRIDE + kb + 16);
    }
#pragma unroll
    for (int ni = 0; ni < 4; ++ni) {
      const int r  = wn * 64 + ni * 16 + (lane & 15);   // lane -> N
      const int kb = (lane >> 4) * 8;                   // K 0-7/16-23 / 8-15/24-31
      b[ni].h[0] = *(const u16x8*)(Bs + r * LDS_STRIDE + kb);
      b[ni].h[1] = *(const u16x8*)(Bs + r * LDS_STRIDE + kb + 16);
    }

#pragma unroll
    for (int mi = 0; mi < 2; ++mi)
#pragma unroll
      for (int ni = 0; ni < 4; ++ni)
        acc[mi][ni] = __builtin_amdgcn_wmma_f32_16x16x32_bf16(
            false, a[mi].v, false, b[ni].v, (short)0, acc[mi][ni], false, false);
    __syncthreads();
  }

  // ---- store: C/D layout: VGPR r, lane l -> M = r + (l>>4)*8, N = l&15 ----
#pragma unroll
  for (int mi = 0; mi < 2; ++mi) {
#pragma unroll
    for (int ni = 0; ni < 4; ++ni) {
      const int n = n0 + wn * 64 + ni * 16 + (lane & 15);
      if (n >= p.N) continue;
#pragma unroll
      for (int r = 0; r < 8; ++r) {
        const int m = m0 + wm * 32 + mi * 16 + (lane >> 4) * 8 + r;
        const float val = acc[mi][ni][r] * p.scale;
        if constexpr (SMODE == 0) {        // head_out[(z>>4)*2048+m][ (z&15)*64+n ]
          const long long row = (long long)(z >> 4) * 2048 + m;
          const long long col = (long long)(z & 15) * 64 + n;
          ((unsigned short*)p.D)[row * 1024 + col] = f2bf(val);
        } else if constexpr (SMODE == 1) { // Q/K: [b,h,q,d] bf16
          const long long bb = m >> 11, q = m & 2047, h = n >> 6, d = n & 63;
          ((unsigned short*)p.D)[(((bb * 16 + h) * 2048) + q) * 64 + d] = f2bf(val);
        } else if constexpr (SMODE == 2) { // V^T: [b,h,d,q] bf16
          const long long bb = m >> 11, q = m & 2047, h = n >> 6, d = n & 63;
          ((unsigned short*)p.D)[(((bb * 16 + h) * 64) + d) * 2048 + q] = f2bf(val);
        } else {                           // fp32 row-major (+ per-z offset)
          ((float*)p.D)[(long long)z * p.dZ + (long long)m * p.ldd + n] = val;
        }
      }
    }
  }
}

// ---------------------------------------------------------------------------
// Row-wise softmax, in place, one 256-thread block per row of `cols` fp32.
// ---------------------------------------------------------------------------
__global__ __launch_bounds__(256) void softmax_rows(float* __restrict__ attn,
                                                    int cols) {
  __shared__ float red[256];
  float* row = attn + (long long)blockIdx.x * cols;
  const int t = threadIdx.x;

  float mx = -3.402823466e38f;
  for (int i = t; i < cols; i += 256) mx = fmaxf(mx, row[i]);
  red[t] = mx;
  __syncthreads();
  for (int s = 128; s > 0; s >>= 1) {
    if (t < s) red[t] = fmaxf(red[t], red[t + s]);
    __syncthreads();
  }
  mx = red[0];
  __syncthreads();

  float sum = 0.f;
  for (int i = t; i < cols; i += 256) {
    const float e = __expf(row[i] - mx);
    row[i] = e;
    sum += e;
  }
  red[t] = sum;
  __syncthreads();
  for (int s = 128; s > 0; s >>= 1) {
    if (t < s) red[t] += red[t + s];
    __syncthreads();
  }
  const float inv = 1.0f / red[0];
  for (int i = t; i < cols; i += 256) row[i] *= inv;
}

// ---------------------------------------------------------------------------
// fp32 -> bf16 cast (RNE), 4 elements per thread.
// ---------------------------------------------------------------------------
__global__ __launch_bounds__(256) void cast_f32_to_bf16(
    const float* __restrict__ src, unsigned short* __restrict__ dst,
    long long n) {
  long long i = ((long long)blockIdx.x * 256 + threadIdx.x) * 4;
  if (i + 3 < n) {
    const float4 f = *(const float4*)(src + i);
    u16x4 v;
    v[0] = f2bf(f.x); v[1] = f2bf(f.y); v[2] = f2bf(f.z); v[3] = f2bf(f.w);
    *(u16x4*)(dst + i) = v;
  } else {
    for (; i < n; ++i) dst[i] = f2bf(src[i]);
  }
}

// ---------------------------------------------------------------------------
// Host-side orchestration.
// ---------------------------------------------------------------------------
extern "C" void kernel_launch(void* const* d_in, const int* in_sizes, int n_in,
                              void* d_out, int out_size, void* d_ws,
                              size_t ws_size, hipStream_t stream) {
  (void)in_sizes; (void)n_in; (void)out_size; (void)ws_size;

  const float* x  = (const float*)d_in[0];
  const float* wq = (const float*)d_in[1];
  const float* wk = (const float*)d_in[2];
  const float* wv = (const float*)d_in[3];
  const float* wo = (const float*)d_in[4];

  constexpr int  Bb = 2, S = 2048, DM = 1024, Hh = 16, DK = 64;
  constexpr long long TOK = (long long)Bb * S;   // 4096
  constexpr long long XE  = TOK * DM;            // 4,194,304 elements
  constexpr long long WE  = (long long)DM * DM;  // 1,048,576 elements

  // Workspace layout (bf16 buffers), ~48 MB total.
  char* ws = (char*)d_ws;
  unsigned short* xb  = (unsigned short*)ws;            // x            [TOK,DM]
  unsigned short* wqb = (unsigned short*)(ws + XE * 2); // W_Q          [DM,DM]
  unsigned short* wkb = wqb + WE;
  unsigned short* wvb = wkb + WE;
  unsigned short* wob = wvb + WE;
  unsigned short* Qh  = wob + WE;                       // Q  [B,H,S,DK]
  unsigned short* Kh  = Qh + XE;                        // K  [B,H,S,DK]
  unsigned short* Vt  = Kh + XE;                        // V^T[B,H,DK,S]
  unsigned short* Ho  = Vt + XE;                        // head_out [TOK,DM]

  float* out  = (float*)d_out;   // [B,S,DM]
  float* attn = out + XE;        // [B,H,S,S]

  // 1) casts
  cast_f32_to_bf16<<<(unsigned)(XE / 1024), 256, 0, stream>>>(x, xb, XE);
  cast_f32_to_bf16<<<(unsigned)(WE / 1024), 256, 0, stream>>>(wq, wqb, WE);
  cast_f32_to_bf16<<<(unsigned)(WE / 1024), 256, 0, stream>>>(wk, wkb, WE);
  cast_f32_to_bf16<<<(unsigned)(WE / 1024), 256, 0, stream>>>(wv, wvb, WE);
  cast_f32_to_bf16<<<(unsigned)(WE / 1024), 256, 0, stream>>>(wo, wob, WE);

  // 2) projections: x @ W^T   (B operand = W row-major [out,in] directly)
  GemmP p{};
  p.A = xb;  p.lda = DM;
  p.ldb = DM; p.M = (int)TOK; p.N = DM; p.K = DM;
  p.aZ = 0;  p.bZ = 0; p.dZ = 0; p.ldd = 0; p.scale = 1.0f;
  const dim3 gproj(DM / 128, (unsigned)(TOK / 128), 1);
  p.B = wqb; p.D = Qh;  gemm_wmma_bf16<0, 1><<<gproj, 256, 0, stream>>>(p);
  p.B = wkb; p.D = Kh;  gemm_wmma_bf16<0, 1><<<gproj, 256, 0, stream>>>(p);
  p.B = wvb; p.D = Vt;  gemm_wmma_bf16<0, 2><<<gproj, 256, 0, stream>>>(p);

  // 3) scores = Q K^T / 8  -> attn region (pre-softmax fp32)
  GemmP sc{};
  sc.A = Qh;   sc.lda = DK; sc.aZ = (long long)S * DK;
  sc.B = Kh;   sc.ldb = DK; sc.bZ = (long long)S * DK;
  sc.D = attn; sc.ldd = S;  sc.dZ = (long long)S * S;
  sc.M = S; sc.N = S; sc.K = DK; sc.scale = 0.125f;
  gemm_wmma_bf16<0, 3><<<dim3(S / 128, S / 128, Bb * Hh), 256, 0, stream>>>(sc);

  // 4) softmax rows, in place
  softmax_rows<<<(unsigned)(Bb * Hh * S), 256, 0, stream>>>(attn, S);

  // 5) head_out = attn @ V   (A = fp32 attn, converted to bf16 while staging)
  GemmP av{};
  av.A = attn; av.lda = S;  av.aZ = (long long)S * S;
  av.B = Vt;   av.ldb = S;  av.bZ = (long long)DK * S;
  av.D = Ho;   av.ldd = 0;  av.dZ = 0;
  av.M = S; av.N = DK; av.K = S; av.scale = 1.0f;
  gemm_wmma_bf16<1, 0><<<dim3(1, S / 128, Bb * Hh), 256, 0, stream>>>(av);

  // 6) out = head_out @ W_O^T  (fp32 output)
  GemmP po{};
  po.A = Ho;  po.lda = DM;
  po.B = wob; po.ldb = DM;
  po.D = out; po.ldd = DM; po.dZ = 0;
  po.M = (int)TOK; po.N = DM; po.K = DM; po.scale = 1.0f;
  gemm_wmma_bf16<0, 3><<<dim3(DM / 128, (unsigned)(TOK / 128), 1), 256, 0,
                         stream>>>(po);
}
